// MultiHeadedSelfAttention_67680094650845
// MI455X (gfx1250) — matrix operand order repaired and checked by reading, not compiled, not run on hardware
//
#include <hip/hip_runtime.h>
#include <hip/hip_bf16.h>

typedef __attribute__((ext_vector_type(16))) _Float16 v16h;
typedef __attribute__((ext_vector_type(8)))  _Float16 v8h;
typedef __attribute__((ext_vector_type(4)))  _Float16 v4h;
typedef __attribute__((ext_vector_type(8)))  float    v8f;

#define B_      4
#define SEQ     2048
#define HID     1024
#define NHEAD   16
#define HDIM    64
#define KSTEP   64

// Flash-attention, wave32 WMMA (gfx1250).
// Block = 256 threads = 8 waves; each wave owns a 16-query tile.
// Block covers 128 consecutive queries of one (batch, head).
// K tile row-major in LDS; V tile stored transposed AND key-permuted by
// sigma^-1(key) = (key&15)*4 + (key>>4) so that P staging writes are b64 and
// all WMMA fragment reads are contiguous b128. (P.V is invariant under a
// common relabeling of the contraction index.)
// All 8 B-fragments per matmul phase are loaded into distinct registers
// BEFORE the WMMAs so ds_load_b128 issue overlaps matrix execution.
__global__ __launch_bounds__(256)
void mhsa_fa_wmma_kernel(const float* __restrict__ q,
                         const float* __restrict__ k,
                         const float* __restrict__ v,
                         const int*   __restrict__ mask,
                         float*       __restrict__ out)
{
    __shared__ _Float16 kt [KSTEP][HDIM + 8];   // [key][d]
    __shared__ _Float16 vtT[HDIM][KSTEP + 8];   // [d][sigma^-1(key)]
    __shared__ _Float16 pb [8][16][KSTEP + 8];  // [wave][row][kappa]

    const int tid  = threadIdx.x;
    const int lane = tid & 31;
    const int wave = tid >> 5;

    const int blk  = blockIdx.x;      // [b(4)][h(16)][qblk(16)]
    const int qblk = blk & 15;
    const int h    = (blk >> 4) & 15;
    const int b    = blk >> 8;

    const int    qbase = qblk * 128 + wave * 16;
    const size_t bq    = (size_t)b * SEQ * HID;
    const int    hoff  = h * HDIM;

    const int arow  = lane & 15;              // A-frag: M = lane&15
    const int abase = (lane >> 4) << 3;       // unused; A-frags also use bkoff (k = 16h + i)
    const int bcol  = lane & 15;              // B/C-frag: N = lane&15
    const int bkoff = (lane >> 4) << 4;       // B-frag K base per lane-half
    const int chalf = (lane >> 4) << 3;       // C-frag row base per lane-half

    // ---- Q tile -> two A-fragments (d 0..31, 32..63); fold softmax scale ----
    v16h qa0, qa1;
    {
        const float* qrow = q + bq + (size_t)(qbase + arow) * HID + hoff;
#pragma unroll
        for (int i = 0; i < 16; ++i) {
            int dk = bkoff + i;
            qa0[i] = (_Float16)(qrow[dk]      * 0.125f);   // 1/sqrt(64)
            qa1[i] = (_Float16)(qrow[32 + dk] * 0.125f);
        }
    }

    float mrun[8], lsum[8];
#pragma unroll
    for (int r = 0; r < 8; ++r) { mrun[r] = -1e30f; lsum[r] = 0.0f; }
    v8f o[4] = {{}, {}, {}, {}};

    // staging geometry: thread -> (key row, 16 contiguous dims)
    const int skey = tid >> 2;                 // 0..63
    const int sd   = (tid & 3) << 4;           // 0,16,32,48
    const int vcol = ((skey & 15) << 2) + (skey >> 4);   // sigma^-1(skey)

    for (int kb = 0; kb < SEQ; kb += KSTEP) {
        __syncthreads();
        {
            const float4* krow = (const float4*)&k[bq + (size_t)(kb + skey) * HID + hoff + sd];
            const float4* vrow = (const float4*)&v[bq + (size_t)(kb + skey) * HID + hoff + sd];
            float4 kf0 = krow[0], kf1 = krow[1], kf2 = krow[2], kf3 = krow[3];
            float4 vf0 = vrow[0], vf1 = vrow[1], vf2 = vrow[2], vf3 = vrow[3];
            v8h kp0 = {(_Float16)kf0.x, (_Float16)kf0.y, (_Float16)kf0.z, (_Float16)kf0.w,
                       (_Float16)kf1.x, (_Float16)kf1.y, (_Float16)kf1.z, (_Float16)kf1.w};
            v8h kp1 = {(_Float16)kf2.x, (_Float16)kf2.y, (_Float16)kf2.z, (_Float16)kf2.w,
                       (_Float16)kf3.x, (_Float16)kf3.y, (_Float16)kf3.z, (_Float16)kf3.w};
            *(v8h*)&kt[skey][sd]     = kp0;
            *(v8h*)&kt[skey][sd + 8] = kp1;
            vtT[sd +  0][vcol] = (_Float16)vf0.x;
            vtT[sd +  1][vcol] = (_Float16)vf0.y;
            vtT[sd +  2][vcol] = (_Float16)vf0.z;
            vtT[sd +  3][vcol] = (_Float16)vf0.w;
            vtT[sd +  4][vcol] = (_Float16)vf1.x;
            vtT[sd +  5][vcol] = (_Float16)vf1.y;
            vtT[sd +  6][vcol] = (_Float16)vf1.z;
            vtT[sd +  7][vcol] = (_Float16)vf1.w;
            vtT[sd +  8][vcol] = (_Float16)vf2.x;
            vtT[sd +  9][vcol] = (_Float16)vf2.y;
            vtT[sd + 10][vcol] = (_Float16)vf2.z;
            vtT[sd + 11][vcol] = (_Float16)vf2.w;
            vtT[sd + 12][vcol] = (_Float16)vf3.x;
            vtT[sd + 13][vcol] = (_Float16)vf3.y;
            vtT[sd + 14][vcol] = (_Float16)vf3.z;
            vtT[sd + 15][vcol] = (_Float16)vf3.w;
        }
        // prefetch next K/V tile rows (global_prefetch_b8)
        if (kb + KSTEP < SEQ) {
            size_t base = bq + (size_t)(kb + KSTEP + (tid & 63)) * HID + hoff;
            if (tid < 64)        __builtin_prefetch(&k[base], 0, 1);
            else if (tid < 128)  __builtin_prefetch(&v[base], 0, 1);
        }
        __syncthreads();

        // ---- mask penalty (hoisted; overlaps with LDS/WMMA) ----
        float pen[4];
#pragma unroll
        for (int nh = 0; nh < 4; ++nh)
            pen[nh] = -10000.0f * (1.0f - (float)mask[b * SEQ + kb + nh * 16 + bcol]);

        // ---- S = Q*K^T : load ALL 8 B-fragments, then issue 8 WMMAs ----
        v16h bk00, bk10, bk20, bk30, bk01, bk11, bk21, bk31;
#pragma unroll
        for (int i = 0; i < 16; ++i) {
            int kk = bkoff + i;
            bk00[i] = kt[ 0 + bcol][kk];
            bk10[i] = kt[16 + bcol][kk];
            bk20[i] = kt[32 + bcol][kk];
            bk30[i] = kt[48 + bcol][kk];
            bk01[i] = kt[ 0 + bcol][32 + kk];
            bk11[i] = kt[16 + bcol][32 + kk];
            bk21[i] = kt[32 + bcol][32 + kk];
            bk31[i] = kt[48 + bcol][32 + kk];
        }
        v8f s[4];
        const v8f zero = {};
        s[0] = __builtin_amdgcn_wmma_f32_16x16x32_f16(false, qa0, false, bk00, (short)0, zero, false, false);
        s[1] = __builtin_amdgcn_wmma_f32_16x16x32_f16(false, qa0, false, bk10, (short)0, zero, false, false);
        s[2] = __builtin_amdgcn_wmma_f32_16x16x32_f16(false, qa0, false, bk20, (short)0, zero, false, false);
        s[3] = __builtin_amdgcn_wmma_f32_16x16x32_f16(false, qa0, false, bk30, (short)0, zero, false, false);
        s[0] = __builtin_amdgcn_wmma_f32_16x16x32_f16(false, qa1, false, bk01, (short)0, s[0], false, false);
        s[1] = __builtin_amdgcn_wmma_f32_16x16x32_f16(false, qa1, false, bk11, (short)0, s[1], false, false);
        s[2] = __builtin_amdgcn_wmma_f32_16x16x32_f16(false, qa1, false, bk21, (short)0, s[2], false, false);
        s[3] = __builtin_amdgcn_wmma_f32_16x16x32_f16(false, qa1, false, bk31, (short)0, s[3], false, false);

        // ---- online softmax per row; stage P with permuted-kappa b64 stores ----
#pragma unroll
        for (int r = 0; r < 8; ++r) {
            float v0 = s[0][r] + pen[0];
            float v1 = s[1][r] + pen[1];
            float v2 = s[2][r] + pen[2];
            float v3 = s[3][r] + pen[3];
            float bm = fmaxf(fmaxf(v0, v1), fmaxf(v2, v3));
            bm = fmaxf(bm, __shfl_xor(bm, 1, 32));
            bm = fmaxf(bm, __shfl_xor(bm, 2, 32));
            bm = fmaxf(bm, __shfl_xor(bm, 4, 32));
            bm = fmaxf(bm, __shfl_xor(bm, 8, 32));
            float mn    = fmaxf(mrun[r], bm);
            float alpha = __expf(mrun[r] - mn);
            float p0 = __expf(v0 - mn);
            float p1 = __expf(v1 - mn);
            float p2 = __expf(v2 - mn);
            float p3 = __expf(v3 - mn);
            mrun[r] = mn;
            lsum[r] = lsum[r] * alpha + (p0 + p1) + (p2 + p3);
            o[0][r] *= alpha; o[1][r] *= alpha; o[2][r] *= alpha; o[3][r] *= alpha;
            v4h pp = {(_Float16)p0, (_Float16)p1, (_Float16)p2, (_Float16)p3};
            *(v4h*)&pb[wave][r + chalf][bcol << 2] = pp;   // kappa = bcol*4+nh
        }

        // ---- P (A-layout, permuted kappa) + ALL 8 V-fragments, then 8 WMMAs ----
        v16h pa0, pa1;
#pragma unroll
        for (int i = 0; i < 16; ++i) {
            int kk = bkoff + i;
            pa0[i] = pb[wave][arow][kk];
            pa1[i] = pb[wave][arow][32 + kk];
        }
        v16h vb00, vb10, vb20, vb30, vb01, vb11, vb21, vb31;
#pragma unroll
        for (int i = 0; i < 16; ++i) {
            int kk = bkoff + i;
            vb00[i] = vtT[ 0 + bcol][kk];
            vb10[i] = vtT[16 + bcol][kk];
            vb20[i] = vtT[32 + bcol][kk];
            vb30[i] = vtT[48 + bcol][kk];
            vb01[i] = vtT[ 0 + bcol][32 + kk];
            vb11[i] = vtT[16 + bcol][32 + kk];
            vb21[i] = vtT[32 + bcol][32 + kk];
            vb31[i] = vtT[48 + bcol][32 + kk];
        }
        o[0] = __builtin_amdgcn_wmma_f32_16x16x32_f16(false, pa0, false, vb00, (short)0, o[0], false, false);
        o[1] = __builtin_amdgcn_wmma_f32_16x16x32_f16(false, pa0, false, vb10, (short)0, o[1], false, false);
        o[2] = __builtin_amdgcn_wmma_f32_16x16x32_f16(false, pa0, false, vb20, (short)0, o[2], false, false);
        o[3] = __builtin_amdgcn_wmma_f32_16x16x32_f16(false, pa0, false, vb30, (short)0, o[3], false, false);
        o[0] = __builtin_amdgcn_wmma_f32_16x16x32_f16(false, pa1, false, vb01, (short)0, o[0], false, false);
        o[1] = __builtin_amdgcn_wmma_f32_16x16x32_f16(false, pa1, false, vb11, (short)0, o[1], false, false);
        o[2] = __builtin_amdgcn_wmma_f32_16x16x32_f16(false, pa1, false, vb21, (short)0, o[2], false, false);
        o[3] = __builtin_amdgcn_wmma_f32_16x16x32_f16(false, pa1, false, vb31, (short)0, o[3], false, false);
    }

    // ---- finalize: reduce row-sums across lanes, normalize, store ----
#pragma unroll
    for (int r = 0; r < 8; ++r) {
        float t = lsum[r];
        t += __shfl_xor(t, 1, 32);
        t += __shfl_xor(t, 2, 32);
        t += __shfl_xor(t, 4, 32);
        t += __shfl_xor(t, 8, 32);
        float inv = 1.0f / t;
        int row = qbase + r + chalf;
        size_t oo = bq + (size_t)row * HID + hoff;
        out[oo + bcol]      = o[0][r] * inv;
        out[oo + 16 + bcol] = o[1][r] * inv;
        out[oo + 32 + bcol] = o[2][r] * inv;
        out[oo + 48 + bcol] = o[3][r] * inv;
    }
}

extern "C" void kernel_launch(void* const* d_in, const int* in_sizes, int n_in,
                              void* d_out, int out_size, void* d_ws, size_t ws_size,
                              hipStream_t stream) {
    const float* q    = (const float*)d_in[0];
    const float* k    = (const float*)d_in[1];
    const float* v    = (const float*)d_in[2];
    const int*   mask = (const int*)d_in[3];
    float*       out  = (float*)d_out;
    (void)in_sizes; (void)n_in; (void)out_size; (void)d_ws; (void)ws_size;

    dim3 grid(B_ * NHEAD * (SEQ / 128));   // 1024 blocks
    dim3 block(256);                        // 8 waves (wave32)
    mhsa_fa_wmma_kernel<<<grid, block, 0, stream>>>(q, k, v, mask, out);
}
